// LorenzFusionPSIWithHooks_37082747633924
// MI455X (gfx1250) — compile-verified, matrix operand order checked by reading, not run on hardware
//
#include <hip/hip_runtime.h>

// ---------------------------------------------------------------------------
// LorenzFusionPSI for MI455X (gfx1250, wave32, WMMA).
//   P0: convert x -> bf16; transpose-convert W_* -> bf16 [N,K]
//   P1: fused 5-way projection GEMM, 32x32 per wave (2x2 WMMA tiles)
//   P2: chunked scan over S (16 chunks of 256)
//   P3: output GEMM with LDS-staged double-buffered B panel
//       (async global->LDS copies when the gfx1250 builtins are available)
// ---------------------------------------------------------------------------

typedef __bf16 bf16;
typedef bf16  bfx4  __attribute__((ext_vector_type(4)));
typedef bf16  bfx8  __attribute__((ext_vector_type(8)));
typedef bf16  bfx16 __attribute__((ext_vector_type(16)));
typedef float f32x4 __attribute__((ext_vector_type(4)));
typedef float f32x8 __attribute__((ext_vector_type(8)));
typedef int   i32x4 __attribute__((ext_vector_type(4)));

constexpr int Bn = 4, Sn = 4096, Dn = 1024;
constexpr int Mn = Bn * Sn;          // 16384 rows (b*s)
constexpr int Fn = 4 * Dn;           // 4096 context features
constexpr int NCHUNK = 16;
constexpr int CHUNK  = Sn / NCHUNK;  // 256
constexpr int BPAD   = 40;           // padded LDS row stride (bf16 elems)

#if defined(__AMDGCN__) && __has_builtin(__builtin_amdgcn_global_load_async_to_lds_b128) && \
    __has_builtin(__builtin_amdgcn_s_wait_asynccnt)
#define USE_ASYNC_LDS 1
#else
#define USE_ASYNC_LDS 0
#endif

__device__ __forceinline__ float sigmoidf_(float x) { return 1.0f / (1.0f + expf(-x)); }

// ---------------- P0: dtype conversion / transpose ----------------

__global__ void k_cvt_bf16(const f32x4* __restrict__ in, bfx4* __restrict__ out, size_t n4) {
  size_t i = (size_t)blockIdx.x * blockDim.x + threadIdx.x;
  if (i < n4) out[i] = __builtin_convertvector(in[i], bfx4);
}

// in: f32 [K,N] row-major, out: bf16 [N,K] row-major
__global__ void k_transpose_bf16(const float* __restrict__ in, bf16* __restrict__ out,
                                 int K, int N) {
  __shared__ float tile[32][33];
  int n0 = blockIdx.x * 32, k0 = blockIdx.y * 32;
  int tx = threadIdx.x, ty = threadIdx.y;
  tile[ty][tx] = in[(size_t)(k0 + ty) * N + (n0 + tx)];
  __syncthreads();
  out[(size_t)(n0 + ty) * K + (k0 + tx)] = (bf16)tile[tx][ty];
}

// ---------------- WMMA fragment loads ---------------------------------------
// ISA 7.12.2 (16-bit A 16x32 / B 32x16 via transposed storage): lane holds
// row/col (L&15); K-halves by L>>4; both halves contiguous -> two b128 loads.
__device__ __forceinline__ bfx16 load_frag(const bf16* __restrict__ p, size_t ld) {
  int lane = threadIdx.x & 31;
  const bf16* row = p + (size_t)(lane & 15) * ld + (lane >> 4) * 8;
  union { bfx16 v; bfx8 h[2]; } u;
  u.h[0] = *(const bfx8*)(row);
  u.h[1] = *(const bfx8*)(row + 16);
  return u.v;
}

__device__ __forceinline__ bfx16 load_frag_lds(const bf16* p) {
  int lane = threadIdx.x & 31;
  const bf16* row = p + (lane & 15) * BPAD + (lane >> 4) * 8;
  union { bfx16 v; bfx8 h[2]; } u;
  u.h[0] = *(const bfx8*)(row);
  u.h[1] = *(const bfx8*)(row + 16);
  return u.v;
}

#define WMMA_BF16(a, b, c) \
  __builtin_amdgcn_wmma_f32_16x16x32_bf16(false, (a), false, (b), (short)0, (c), false, false)

// ---------------- P1: fused 5-projection GEMM, 32x32 per wave ---------------

__global__ void __launch_bounds__(256) k_proj5(
    const bf16* __restrict__ xb,   // [Mn, Dn] bf16
    const bf16* __restrict__ wt,   // 5 x [Dn, Dn] bf16 (transposed weights)
    const float* __restrict__ b_om, const float* __restrict__ b_mg,
    const float* __restrict__ b_ph, const float* __restrict__ b_gt,
    const float* __restrict__ b_q,  const float* __restrict__ iscale,
    float* __restrict__ gated, float* __restrict__ magn,
    float* __restrict__ phii,  float* __restrict__ qoff) {
  const int ntn = Dn / 32;                         // 32 N-supertiles
  int wave = threadIdx.x >> 5;
  int lane = threadIdx.x & 31;
  int tile = blockIdx.x * 8 + wave;                // 16384 tiles total
  int tm = tile / ntn, tn = tile % ntn;
  int m0 = tm * 32, n0 = tn * 32;

  f32x8 acc[5][4] = {};                            // [weight][mt*2+nt]
  const bf16* arow0 = xb + (size_t)m0 * Dn;
  const bf16* arow1 = xb + (size_t)(m0 + 16) * Dn;
  for (int k0 = 0; k0 < Dn; k0 += 32) {
    __builtin_prefetch(arow0 + k0 + 128, 0, 3);
    bfx16 a0 = load_frag(arow0 + k0, Dn);
    bfx16 a1 = load_frag(arow1 + k0, Dn);
#pragma unroll
    for (int w = 0; w < 5; w++) {
      const bf16* bb = wt + (size_t)w * Dn * Dn + (size_t)n0 * Dn + k0;
      bfx16 b0 = load_frag(bb, Dn);
      bfx16 b1 = load_frag(bb + (size_t)16 * Dn, Dn);
      acc[w][0] = WMMA_BF16(a0, b0, acc[w][0]);
      acc[w][1] = WMMA_BF16(a0, b1, acc[w][1]);
      acc[w][2] = WMMA_BF16(a1, b0, acc[w][2]);
      acc[w][3] = WMMA_BF16(a1, b1, acc[w][3]);
    }
  }

#pragma unroll
  for (int nt = 0; nt < 2; nt++) {
    int n = n0 + nt * 16 + (lane & 15);
    float sc = fabsf(iscale[n]);
    float bo = b_om[n], bm = b_mg[n], bp = b_ph[n], bg = b_gt[n], bq = b_q[n];
#pragma unroll
    for (int mt = 0; mt < 2; mt++) {
      int mb = m0 + mt * 16 + ((lane >> 4) ? 8 : 0);
      int t = mt * 2 + nt;
#pragma unroll
      for (int j = 0; j < 8; j++) {
        size_t idx = (size_t)(mb + j) * Dn + n;
        float om = acc[0][t][j] + bo;
        float mg = 5.0f * sigmoidf_(acc[1][t][j] + bm);
        float ph = acc[2][t][j] + bp;
        float gt = sigmoidf_(acc[3][t][j] + bg);
        float qv = acc[4][t][j] + bq;
        gated[idx] = gt * om * sc;
        magn[idx]  = mg;
        phii[idx]  = ph;
        qoff[idx]  = qv;
      }
    }
  }
}

// ---------------- P2: chunked scans ----------------------------------------

__device__ __forceinline__ void scan_idx(int t, int& b, int& c, int& d, size_t& base) {
  d = t % Dn;
  c = (t / Dn) % NCHUNK;
  b = t / (Dn * NCHUNK);
  base = ((size_t)(b * Sn + c * CHUNK)) * Dn + d;
}

__global__ void k_scan_sums(const float* __restrict__ gated, const float* __restrict__ magn,
                            float* __restrict__ sPhi, float* __restrict__ sMag) {
  int t = blockIdx.x * blockDim.x + threadIdx.x;   // Bn*NCHUNK*Dn threads
  int b, c, d; size_t base; scan_idx(t, b, c, d, base);
  float sp = 0.f, sm = 0.f;
  for (int s = 0; s < CHUNK; s++) {
    sp += gated[base + (size_t)s * Dn];
    sm += magn [base + (size_t)s * Dn];
  }
  sPhi[t] = sp; sMag[t] = sm;
}

__global__ void k_prefix2(float* __restrict__ a, float* __restrict__ b2) {
  int t = blockIdx.x * blockDim.x + threadIdx.x;   // Bn*Dn threads
  int d = t % Dn, b = t / Dn;
  float ra = 0.f, rb = 0.f;
  for (int c = 0; c < NCHUNK; c++) {
    size_t i = ((size_t)(b * NCHUNK + c)) * Dn + d;
    float ta = a[i], tb = b2[i];
    a[i] = ra; b2[i] = rb;
    ra += ta; rb += tb;
  }
}

__global__ void k_scan_reim(const float* __restrict__ gated, const float* __restrict__ phii,
                            const float* __restrict__ magn,  const float* __restrict__ x,
                            const float* __restrict__ sPhi,
                            float* __restrict__ sRe, float* __restrict__ sIm) {
  int t = blockIdx.x * blockDim.x + threadIdx.x;
  int b, c, d; size_t base; scan_idx(t, b, c, d, base);
  float P = sPhi[t];
  float re = 0.f, im = 0.f;
  for (int s = 0; s < CHUNK; s++) {
    size_t i = base + (size_t)s * Dn;
    float g = gated[i]; P += g;
    float phi = phii[i] + P;
    float wc = magn[i] * x[i];
    float sp, cp; sincosf(phi, &sp, &cp);
    re += wc * cp; im += wc * sp;
  }
  sRe[t] = re; sIm[t] = im;
}

__global__ void k_scan_final(const float* __restrict__ gated, const float* __restrict__ phii,
                             const float* __restrict__ magn,  const float* __restrict__ qoff,
                             const float* __restrict__ x,
                             const float* __restrict__ sPhi, const float* __restrict__ sMag,
                             const float* __restrict__ sRe,  const float* __restrict__ sIm,
                             bf16* __restrict__ ctx) {        // [Mn, Fn]
  int t = blockIdx.x * blockDim.x + threadIdx.x;
  int b, c, d; size_t base; scan_idx(t, b, c, d, base);
  float P = sPhi[t], Mg = sMag[t], Re = sRe[t], Im = sIm[t];
  for (int s = 0; s < CHUNK; s++) {
    size_t i = base + (size_t)s * Dn;
    float g = gated[i]; P += g;
    float phi = phii[i] + P;
    float xv = x[i], mg = magn[i];
    float wc = mg * xv;
    float sp, cp; sincosf(phi, &sp, &cp);
    Re += wc * cp; Im += wc * sp; Mg += mg;
    float rs = 1.0f / sqrtf(Mg + 1e-8f);
    float mr = Re * rs, mi = Im * rs;
    float pq = phi + qoff[i];
    float sq, cq; sincosf(pq, &sq, &cq);
    float rr = mr * cq + mi * sq;
    float ri = mi * cq - mr * sq;
    size_t cb = ((size_t)(b * Sn + c * CHUNK + s)) * Fn;
    ctx[cb + d]          = (bf16)(xv * cp);
    ctx[cb + Dn + d]     = (bf16)(xv * sp);
    ctx[cb + 2 * Dn + d] = (bf16)rr;
    ctx[cb + 3 * Dn + d] = (bf16)ri;
  }
}

// ---------------- P3: output GEMM + residual (LDS-staged B panel) ----------

__device__ __forceinline__ void stage_b128(const bf16* g, bf16* l) {
#if USE_ASYNC_LDS
  __builtin_amdgcn_global_load_async_to_lds_b128(
      (__attribute__((address_space(1))) i32x4*)(g),
      (__attribute__((address_space(3))) i32x4*)(l), 0, 0);
#else
  *(bfx8*)l = *(const bfx8*)g;
#endif
}

__device__ __forceinline__ void stage_wait() {
#if USE_ASYNC_LDS
  __builtin_amdgcn_s_wait_asynccnt(0);
#endif
  __syncthreads();
}

__global__ void __launch_bounds__(256) k_outproj(
    const bf16* __restrict__ ctx,   // [Mn, Fn]
    const bf16* __restrict__ wot,   // [Dn, Fn]  (W_out^T)
    const float* __restrict__ x, const float* __restrict__ b_out,
    float* __restrict__ out) {
  __shared__ bf16 Bs[2][64][BPAD];          // double-buffered 64x32 B panel

  int wave = threadIdx.x >> 5;
  int lane = threadIdx.x & 31;
  int bm = blockIdx.x >> 4;                 // Mn/128 = 128
  int bnb = blockIdx.x & 15;                // Dn/64  = 16
  int m_base = bm * 128, n_base = bnb * 64;
  int wm = wave >> 1, wn = wave & 1;
  int m0  = m_base + wm * 32;
  int ln0 = wn * 32;                        // wave's local N rows in panel

  // panel staging: 256 threads x one b128 each = 64 rows x 64 bytes
  int srow = threadIdx.x >> 2;
  int sseg = (threadIdx.x & 3) * 8;
  const bf16* bsrc = wot + (size_t)(n_base + srow) * Fn + sseg;

  const bf16* arow0 = ctx + (size_t)m0 * Fn;
  const bf16* arow1 = ctx + (size_t)(m0 + 16) * Fn;

  f32x8 acc[4] = {};
  stage_b128(bsrc, &Bs[0][srow][sseg]);
  stage_wait();

  int buf = 0;
  for (int k0 = 0; k0 < Fn; k0 += 32) {
    if (k0 + 32 < Fn) stage_b128(bsrc + k0 + 32, &Bs[buf ^ 1][srow][sseg]);
    __builtin_prefetch(arow0 + k0 + 128, 0, 3);
    __builtin_prefetch(arow1 + k0 + 128, 0, 3);
    bfx16 a0 = load_frag(arow0 + k0, Fn);
    bfx16 a1 = load_frag(arow1 + k0, Fn);
    bfx16 b0 = load_frag_lds(&Bs[buf][ln0][0]);
    bfx16 b1 = load_frag_lds(&Bs[buf][ln0 + 16][0]);
    acc[0] = WMMA_BF16(a0, b0, acc[0]);
    acc[1] = WMMA_BF16(a0, b1, acc[1]);
    acc[2] = WMMA_BF16(a1, b0, acc[2]);
    acc[3] = WMMA_BF16(a1, b1, acc[3]);
    stage_wait();
    buf ^= 1;
  }

  int n0w = n_base + wn * 32;
  float bo0 = b_out[n0w + (lane & 15)];
  float bo1 = b_out[n0w + 16 + (lane & 15)];
#pragma unroll
  for (int mt = 0; mt < 2; mt++) {
    int mb = m0 + mt * 16 + ((lane >> 4) ? 8 : 0);
#pragma unroll
    for (int j = 0; j < 8; j++) {
      size_t i0 = (size_t)(mb + j) * Dn + n0w + (lane & 15);
      size_t i1 = i0 + 16;
      out[i0] = x[i0] + bo0 + acc[mt * 2 + 0][j];
      out[i1] = x[i1] + bo1 + acc[mt * 2 + 1][j];
    }
  }
}

// ---------------- host launcher --------------------------------------------

extern "C" void kernel_launch(void* const* d_in, const int* in_sizes, int n_in,
                              void* d_out, int out_size, void* d_ws, size_t ws_size,
                              hipStream_t stream) {
  (void)in_sizes; (void)n_in; (void)out_size; (void)ws_size;

  const float* x       = (const float*)d_in[0];
  const float* W5[5]   = {(const float*)d_in[1], (const float*)d_in[3],
                          (const float*)d_in[5], (const float*)d_in[7],
                          (const float*)d_in[9]};
  const float* b_om    = (const float*)d_in[2];
  const float* b_mg    = (const float*)d_in[4];
  const float* b_ph    = (const float*)d_in[6];
  const float* b_gt    = (const float*)d_in[8];
  const float* b_q     = (const float*)d_in[10];
  const float* iscale  = (const float*)d_in[11];
  const float* W_out   = (const float*)d_in[12];
  const float* b_out   = (const float*)d_in[13];
  float* out = (float*)d_out;

  auto al = [](size_t v) { return (v + 255) & ~(size_t)255; };
  char* w = (char*)d_ws;
  size_t off = 0;
  bf16* xb    = (bf16*)(w + off); off += al((size_t)Mn * Dn * 2);
  bf16* wt5   = (bf16*)(w + off); off += al((size_t)5 * Dn * Dn * 2);
  bf16* wot   = (bf16*)(w + off); off += al((size_t)Dn * Fn * 2);
  float* gated = (float*)(w + off); off += al((size_t)Mn * Dn * 4);
  float* magn  = (float*)(w + off); off += al((size_t)Mn * Dn * 4);
  float* phii  = (float*)(w + off); off += al((size_t)Mn * Dn * 4);
  float* qoff  = (float*)(w + off); off += al((size_t)Mn * Dn * 4);
  bf16* ctx   = (bf16*)(w + off);  off += al((size_t)Mn * Fn * 2);
  float* sPhi = (float*)(w + off); off += al((size_t)Bn * NCHUNK * Dn * 4);
  float* sMag = (float*)(w + off); off += al((size_t)Bn * NCHUNK * Dn * 4);
  float* sRe  = (float*)(w + off); off += al((size_t)Bn * NCHUNK * Dn * 4);
  float* sIm  = (float*)(w + off); off += al((size_t)Bn * NCHUNK * Dn * 4);

  // P0: conversions
  {
    size_t n4 = (size_t)Mn * Dn / 4;
    k_cvt_bf16<<<(unsigned)(n4 / 256), 256, 0, stream>>>((const f32x4*)x, (bfx4*)xb, n4);
  }
  {
    dim3 blk(32, 32);
    dim3 grd(Dn / 32, Dn / 32);
    for (int i = 0; i < 5; i++)
      k_transpose_bf16<<<grd, blk, 0, stream>>>(W5[i], wt5 + (size_t)i * Dn * Dn, Dn, Dn);
    dim3 grdo(Dn / 32, Fn / 32);
    k_transpose_bf16<<<grdo, blk, 0, stream>>>(W_out, wot, Fn, Dn);
  }

  // P1: fused projections (32x32 tiles per wave)
  {
    int tiles = (Mn / 32) * (Dn / 32);   // 16384
    k_proj5<<<tiles / 8, 256, 0, stream>>>(xb, wt5, b_om, b_mg, b_ph, b_gt, b_q,
                                           iscale, gated, magn, phii, qoff);
  }

  // P2: chunked scan
  {
    int nt = Bn * NCHUNK * Dn;           // 65536
    k_scan_sums<<<nt / 256, 256, 0, stream>>>(gated, magn, sPhi, sMag);
    k_prefix2<<<(Bn * Dn) / 256, 256, 0, stream>>>(sPhi, sMag);
    k_scan_reim<<<nt / 256, 256, 0, stream>>>(gated, phii, magn, x, sPhi, sRe, sIm);
    k_prefix2<<<(Bn * Dn) / 256, 256, 0, stream>>>(sRe, sIm);
    k_scan_final<<<nt / 256, 256, 0, stream>>>(gated, phii, magn, qoff, x,
                                               sPhi, sMag, sRe, sIm, ctx);
  }

  // P3: output projection + residual (128x64 block tile, LDS B panel)
  {
    int blocks = (Mn / 128) * (Dn / 64);  // 2048
    k_outproj<<<blocks, 256, 0, stream>>>(ctx, wot, x, b_out, out);
  }
}
